// GraphAttention_77180562309458
// MI455X (gfx1250) — compile-verified, matrix operand order checked by reading, not run on hardware
//
#include <hip/hip_runtime.h>
#include <hip/hip_bf16.h>

typedef __attribute__((ext_vector_type(16))) __bf16 v16bf;
typedef __attribute__((ext_vector_type(8)))  float  v8f;

#define Bn 4096
#define Nn 4096
#define Fn 256
#define Fo 64
#define Hn 8

#define LOG2E 1.44269504088896340736f

// ---------------------------------------------------------------------------
// Kernel 1: v2[h][f] = sum_o kernels[h][f][o] * a2[h][o]; kernels -> bf16.
// (a1/s1 are exactly cancelled by softmax shift-invariance and are dropped.)
// ---------------------------------------------------------------------------
__global__ void prep_kernel(const float* __restrict__ kernels,
                            const float* __restrict__ att,   // [H][2*Fo]
                            float* __restrict__ v2,
                            __bf16* __restrict__ Kb) {
  int h = blockIdx.x;         // 8
  int f = threadIdx.x;        // 256
  __shared__ float a2s[Fo];
  if (threadIdx.x < Fo) a2s[threadIdx.x] = att[h * 2 * Fo + Fo + threadIdx.x];
  __syncthreads();
  const float* kr = kernels + ((size_t)h * Fn + f) * Fo;
  __bf16* kb = Kb + ((size_t)h * Fn + f) * Fo;
  float d2 = 0.f;
#pragma unroll 8
  for (int o = 0; o < Fo; ++o) {
    float kv = kr[o];
    d2 += kv * a2s[o];
    kb[o] = (__bf16)kv;
  }
  v2[h * Fn + f] = d2;
}

// ---------------------------------------------------------------------------
// Kernel 2: s2[h][n] = G[n,:].v2[h,:]
// ---------------------------------------------------------------------------
__global__ void s2_kernel(const float* __restrict__ G,
                          const float* __restrict__ v2,
                          float* __restrict__ s2) {
  int idx = blockIdx.x * 256 + threadIdx.x;   // H*N = 32768 threads
  int h = idx >> 12;                          // uniform per block (256 | 4096)
  int n = idx & 4095;
  __shared__ float vs[Fn];
  vs[threadIdx.x] = v2[h * Fn + threadIdx.x];
  __syncthreads();
  const float* row = G + (size_t)n * Fn;
  float d = 0.f;
#pragma unroll 8
  for (int f = 0; f < Fn; ++f) d += row[f] * vs[f];
  s2[h * Nn + n] = d;
}

// ---------------------------------------------------------------------------
// Kernel 3: per-head fixed softmax shift.  M[h] = max_n s2[h][n];
// rewrite s2 in place as (s2 - M) * log2(e)  (ready for v_exp = exp2).
// Since A in [0,1), M is within 1.0 of every true row max -> numerically safe.
// ---------------------------------------------------------------------------
__global__ void maxsub_kernel(float* __restrict__ s2) {
  int h = blockIdx.x;
  float* p = s2 + h * Nn;
  int t = threadIdx.x;
  float mx = -1e30f;
#pragma unroll 4
  for (int i = t; i < Nn; i += 256) mx = fmaxf(mx, p[i]);
  __shared__ float red[256];
  red[t] = mx;
  __syncthreads();
  for (int s = 128; s > 0; s >>= 1) {
    if (t < s) red[t] = fmaxf(red[t], red[t + s]);
    __syncthreads();
  }
  float M = red[0];
#pragma unroll 4
  for (int i = t; i < Nn; i += 256) p[i] = (p[i] - M) * LOG2E;
}

// ---------------------------------------------------------------------------
// Kernel 4 (generic): swizzle a row-major bf16 [H][rows][64] matrix into
// WMMA B-operand lane order: per (h, c, t) a 1024B block; lane L owns 32B.
// Lane L, col = L&15, half = L>>4; its 16 k's are {8h+0..7} U {16+8h+0..7}.
// ---------------------------------------------------------------------------
__global__ void bswz_kernel(const __bf16* __restrict__ src,
                            __bf16* __restrict__ dst,
                            int cmask, int cshift) {
  int tid = blockIdx.x * 256 + threadIdx.x;
  int L = tid & 31;
  int t = (tid >> 5) & 3;
  int rest = tid >> 7;
  int c = rest & cmask;
  int h = rest >> cshift;
  int col = L & 15, half = L >> 4;
  int rows = (cmask + 1) * 32;
  __bf16* d = dst + ((((size_t)h * (cmask + 1) + c) * 4 + t) * 32 + L) * 16;
  const __bf16* s = src + ((size_t)h * rows + c * 32) * Fo + t * 16 + col;
#pragma unroll
  for (int i = 0; i < 8; ++i) {
    d[i]     = s[(size_t)(8 * half + i) * Fo];
    d[8 + i] = s[(size_t)(16 + 8 * half + i) * Fo];
  }
}

// ---------------------------------------------------------------------------
// Kernel 5: WG[h] = G @ kernels[h]  via v_wmma_f32_16x16x32_bf16.
// B operand read pre-swizzled (contiguous b128 pairs per lane).
// ---------------------------------------------------------------------------
__global__ void wg_gemm_kernel(const float* __restrict__ G,
                               const __bf16* __restrict__ Kswz,
                               __bf16* __restrict__ WGb) {
  int lane = threadIdx.x & 31;
  int wave = threadIdx.x >> 5;
  int wglobal = blockIdx.x * 8 + wave;     // 2048 waves
  int h  = wglobal >> 8;
  int n0 = (wglobal & 255) * 16;
  int col  = lane & 15;
  int half = lane >> 4;
  const float*  grow  = G + (size_t)(n0 + col) * Fn;
  const __bf16* kbase = Kswz + (size_t)h * (8 * 4 * 512) + (size_t)lane * 16;

  v8f acc[4] = {v8f{0}, v8f{0}, v8f{0}, v8f{0}};

  for (int c = 0; c < 8; ++c) {
    int kk = c * 32;
    int k0 = kk + 8 * half;
    int k1 = kk + 16 + 8 * half;
    float av[16];
    *(float4*)(av + 0)  = *(const float4*)(grow + k0);
    *(float4*)(av + 4)  = *(const float4*)(grow + k0 + 4);
    *(float4*)(av + 8)  = *(const float4*)(grow + k1);
    *(float4*)(av + 12) = *(const float4*)(grow + k1 + 4);
    v16bf a;
#pragma unroll
    for (int i = 0; i < 16; ++i) a[i] = (__bf16)av[i];
    const __bf16* kc = kbase + (size_t)c * 2048;
#pragma unroll
    for (int t = 0; t < 4; ++t) {
      v16bf bv = *(const v16bf*)(kc + (size_t)t * 512);
      acc[t] = __builtin_amdgcn_wmma_f32_16x16x32_bf16(
          false, a, false, bv, (short)0, acc[t], false, false);
    }
  }
#pragma unroll
  for (int t = 0; t < 4; ++t)
#pragma unroll
    for (int r = 0; r < 8; ++r) {
      int n = n0 + r + 8 * half;
      int o = t * 16 + col;
      WGb[((size_t)h * Nn + n) * Fo + o] = (__bf16)acc[t][r];
    }
}

// ---------------------------------------------------------------------------
// Kernel 6: streaming softmax-attention with FIXED per-head shift.
// Block = 16 b-rows, wave w = head w. Chunk over N in steps of 32.
// p = exp2( fma(A, -log2e, s2') )  -- one FMA + one v_exp per score.
// No running max / no accumulator rescale (provably safe, see maxsub).
// ---------------------------------------------------------------------------
__global__ void flash_kernel(const float* __restrict__ A,
                             const float* __restrict__ s2m,   // (s2-M)*log2e
                             const __bf16* __restrict__ WGswz,
                             float* __restrict__ out) {
  int lane = threadIdx.x & 31;
  int h    = threadIdx.x >> 5;          // wave id == head
  int b0   = blockIdx.x * 16;
  int col = lane & 15, half = lane >> 4;

  const float*  Arow  = A + (size_t)(b0 + col) * Nn;
  const float*  s2h   = s2m + h * Nn;
  const __bf16* wbase = WGswz + (size_t)h * (128 * 4 * 512) + (size_t)lane * 16;

  v8f acc[4] = {v8f{0}, v8f{0}, v8f{0}, v8f{0}};
  float l = 0.f;

  for (int c = 0; c < 128; ++c) {
    int n0 = c * 32;
    int k0 = n0 + 8 * half;
    int k1 = n0 + 16 + 8 * half;
    float av[16], tv[16];
    *(float4*)(av + 0)  = *(const float4*)(Arow + k0);
    *(float4*)(av + 4)  = *(const float4*)(Arow + k0 + 4);
    *(float4*)(av + 8)  = *(const float4*)(Arow + k1);
    *(float4*)(av + 12) = *(const float4*)(Arow + k1 + 4);
    *(float4*)(tv + 0)  = *(const float4*)(s2h + k0);
    *(float4*)(tv + 4)  = *(const float4*)(s2h + k0 + 4);
    *(float4*)(tv + 8)  = *(const float4*)(s2h + k1);
    *(float4*)(tv + 12) = *(const float4*)(s2h + k1 + 4);

    // prefetch next chunk of A (global_prefetch_b8)
    __builtin_prefetch(Arow + k0 + 32, 0, 1);

    float sum = 0.f;
    v16bf p;
#pragma unroll
    for (int i = 0; i < 16; ++i) {
      float e = __builtin_exp2f(__builtin_fmaf(av[i], -LOG2E, tv[i]));
      sum += e;
      p[i] = (__bf16)e;
    }
    l += sum;

    const __bf16* wc = wbase + (size_t)c * 2048;
#pragma unroll
    for (int t = 0; t < 4; ++t) {
      v16bf bv = *(const v16bf*)(wc + (size_t)t * 512);
      acc[t] = __builtin_amdgcn_wmma_f32_16x16x32_bf16(
          false, p, false, bv, (short)0, acc[t], false, false);
    }
  }

  // combine the two half-lane partial sums of each row
  l += __shfl_xor(l, 16);

  // epilogue: divide by l, ReLU, store f32 out[b][h*64+o]
#pragma unroll
  for (int r = 0; r < 8; ++r) {
    float lr  = __shfl(l, r + 8 * half);
    float inv = 1.0f / lr;
    int b = b0 + r + 8 * half;
    float* orow = out + (size_t)b * (Hn * Fo) + h * Fo;
#pragma unroll
    for (int t = 0; t < 4; ++t) {
      float v = acc[t][r] * inv;
      orow[t * 16 + col] = fmaxf(v, 0.f);
    }
  }
}

// ---------------------------------------------------------------------------
extern "C" void kernel_launch(void* const* d_in, const int* in_sizes, int n_in,
                              void* d_out, int out_size, void* d_ws, size_t ws_size,
                              hipStream_t stream) {
  const float* G       = (const float*)d_in[1];
  const float* A       = (const float*)d_in[2];
  const float* kernels = (const float*)d_in[3];
  const float* att     = (const float*)d_in[4];
  float* out = (float*)d_out;

  char* ws = (char*)d_ws;
  __bf16* Kb    = (__bf16*)(ws);                                   // 256 KB
  __bf16* Kswz  = (__bf16*)(ws + 256 * 1024);                      // 256 KB
  float*  v2    = (float*)(ws + 512 * 1024);                       //   8 KB
  float*  s2    = (float*)(ws + 520 * 1024);                       // 128 KB
  __bf16* WGb   = (__bf16*)(ws + 1024 * 1024);                     //   4 MB
  __bf16* WGswz = (__bf16*)(ws + 5u * 1024 * 1024);                //   4 MB

  prep_kernel   <<<Hn,  256, 0, stream>>>(kernels, att, v2, Kb);
  s2_kernel     <<<128, 256, 0, stream>>>(G, v2, s2);
  maxsub_kernel <<<Hn,  256, 0, stream>>>(s2);
  bswz_kernel   <<<32,  256, 0, stream>>>(Kb, Kswz, 7, 3);          // kernels: 8 chunks
  wg_gemm_kernel<<<256, 256, 0, stream>>>(G, Kswz, WGb);
  bswz_kernel   <<<512, 256, 0, stream>>>(WGb, WGswz, 127, 7);      // WG: 128 chunks
  flash_kernel  <<<256, 256, 0, stream>>>(A, s2, WGswz, out);
}
